// VolumnRenderer_12378095747393
// MI455X (gfx1250) — matrix operand order repaired and checked by reading, not compiled, not run on hardware
//
#include <hip/hip_runtime.h>
#include <hip/hip_bf16.h>
#include <math.h>

typedef _Float16 h8   __attribute__((ext_vector_type(8)));
typedef _Float16 v16h __attribute__((ext_vector_type(16)));
typedef float    v8f  __attribute__((ext_vector_type(8)));

#define NRAYS 8192
#define NS 128
#define GRES 128
#define NBLK 2048
#define RPB (NRAYS / NBLK)              // 4 rays per block
#define AMIN_F (-1.5f)
#define AMAX_F (1.5f)
#define STEP_F (3.0f * 1.7320508075688772f / 128.0f)

#define WMMA_F16(A, B, C) \
  __builtin_amdgcn_wmma_f32_16x16x32_f16(false, (A), false, (B), (short)0, (C), false, false)

// Phases A..E only touch wave-private LDS staging (+ read-only weights).
// LDS ops within a wave are processed in order (CDNA5 ISA, DScnt), so a
// zero-instruction compiler fence is sufficient — no block barrier needed.
#define WAVE_SYNC() asm volatile("" ::: "memory")

struct __align__(64) SM {
  _Float16 W1t[64][32];    // W1 transposed, K padded 27->32
  _Float16 W2t[64][64];
  _Float16 Wct[64][64];
  _Float16 Wst[16][64];    // Ws padded to 16 output cols (col 0 live)
  _Float16 Wrt[16][64];    // Wr padded to 16 output cols (cols 0..2 live)
  _Float16 h1[8][16][64];  // per-wave staging (h1, later reused for g)
  union {
    _Float16 pe[8][16][32];  // posenc staging (dead after layer 1)
    _Float16 h2[8][16][64];
  } u;
  float b1f[64];
  float b2f[64];
  float dfeat[64];         // bc + pe_d @ Wd (per ray)
  float sig[NS];
  float tv[NS];
  float vld[NS];
  float rgbb[NS][3];
};

// A-fragment (16x32 f16, row-major in LDS): lane L -> row L, K k0..k0+15;
// lane L+16 -> row L, K k0+16..k0+31
__device__ __forceinline__ v16h lds_load_A(const _Float16* buf, int stride, int k0, int lane) {
  int row = lane & 15;
  int off = k0 + ((lane & 16) ? 16 : 0);
  const _Float16* p = buf + row * stride + off;
  union { v16h v; h8 h[2]; } u;
  u.h[0] = *(const h8*)(p);
  u.h[1] = *(const h8*)(p + 8);
  return u.v;
}

// B-fragment (32x16 f16) from transposed weights Wt[n][k]:
// lane n<16 -> col n0+n, K k0..k0+15 ; lane n+16 -> col n0+n, K k0+16..k0+31
__device__ __forceinline__ v16h lds_load_B(const _Float16* wt, int stride, int n0, int k0, int lane) {
  int col = lane & 15;
  int off = k0 + ((lane & 16) ? 16 : 0);
  return *(const v16h*)(wt + (n0 + col) * stride + off);
}

__global__ __launch_bounds__(256) void nerf_render_kernel(
    const float* __restrict__ rays_o, const float* __restrict__ viewdirs,
    const unsigned char* __restrict__ occ,
    const float* __restrict__ W1, const float* __restrict__ b1,
    const float* __restrict__ W2, const float* __restrict__ b2,
    const float* __restrict__ Ws, const float* __restrict__ bs,
    const float* __restrict__ Wc, const float* __restrict__ Wd,
    const float* __restrict__ bc, const float* __restrict__ Wr,
    const float* __restrict__ br, float* __restrict__ out)
{
  __shared__ SM sm;
  const int tid  = threadIdx.x;
  const int wave = tid >> 5;
  const int lane = tid & 31;

  // ---- Stage weights (f32 -> f16, transposed) into LDS once per block ----
  for (int i = tid; i < 64 * 32; i += 256) {
    int n = i >> 5, k = i & 31;
    sm.W1t[n][k] = (_Float16)(k < 27 ? W1[k * 64 + n] : 0.0f);
  }
  for (int i = tid; i < 64 * 64; i += 256) {
    int n = i >> 6, k = i & 63;
    sm.W2t[n][k] = (_Float16)W2[k * 64 + n];
    sm.Wct[n][k] = (_Float16)Wc[k * 64 + n];
  }
  for (int i = tid; i < 16 * 64; i += 256) {
    int n = i >> 6, k = i & 63;
    sm.Wst[n][k] = (_Float16)(n == 0 ? Ws[k] : 0.0f);
    sm.Wrt[n][k] = (_Float16)(n < 3 ? Wr[k * 3 + n] : 0.0f);
  }
  if (tid < 64) { sm.b1f[tid] = b1[tid]; sm.b2f[tid] = b2[tid]; }
  const float bsv = bs[0];

  for (int rr = 0; rr < RPB; ++rr) {
    const int ray = blockIdx.x * RPB + rr;

    // ---- Ray setup (every thread, cheap & cached) ----
    const float ox = rays_o[ray * 3 + 0], oy = rays_o[ray * 3 + 1], oz = rays_o[ray * 3 + 2];
    const float dx = viewdirs[ray * 3 + 0], dy = viewdirs[ray * 3 + 1], dz = viewdirs[ray * 3 + 2];
    float tn = 0.0f, tf = 3.402823e38f;
    {
      float o3[3] = {ox, oy, oz}, d3[3] = {dx, dy, dz};
#pragma unroll
      for (int i = 0; i < 3; ++i) {
        float d = d3[i];
        float sd = (fabsf(d) < 1e-8f) ? 1e-8f : d;
        float t1 = (AMIN_F - o3[i]) / sd;
        float t2 = (AMAX_F - o3[i]) / sd;
        tn = fmaxf(tn, fminf(t1, t2));
        tf = fminf(tf, fmaxf(t1, t2));
      }
    }
    // per-ray direction feature: dfeat[j] = bc[j] + pe_d @ Wd
    if (tid < 64) {
      float ped[15];
      ped[0] = dx; ped[1] = dy; ped[2] = dz;
      float dv[3] = {dx, dy, dz};
#pragma unroll
      for (int i = 0; i < 3; ++i) {
#pragma unroll
        for (int l = 0; l < 2; ++l) {
          float a = dv[i] * (float)(1 << l);
          ped[3 + i * 4 + l]     = __sinf(a);
          ped[3 + i * 4 + 2 + l] = __cosf(a);
        }
      }
      float acc = bc[tid];
#pragma unroll
      for (int k = 0; k < 15; ++k) acc += ped[k] * Wd[k * 64 + tid];
      sm.dfeat[tid] = acc;
    }
    __syncthreads();  // cross-wave: weights (iter 0) + dfeat ready

    // ---- Phase A: sample points, occupancy gather, posenc -> LDS (A layout source) ----
    if (lane < 16) {
      int s = wave * 16 + lane;
      float tval = tn + ((float)s + 0.5f) * STEP_F;
      float px = ox + dx * tval, py = oy + dy * tval, pz = oz + dz * tval;
      const float scale = (float)GRES / (AMAX_F - AMIN_F);
      float cx = (px - AMIN_F) * scale, cy = (py - AMIN_F) * scale, cz = (pz - AMIN_F) * scale;
      bool inbox = (cx >= 0.f) && (cx < (float)GRES) && (cy >= 0.f) && (cy < (float)GRES) &&
                   (cz >= 0.f) && (cz < (float)GRES);
      int ix = (int)floorf(cx); ix = ix < 0 ? 0 : (ix > GRES - 1 ? GRES - 1 : ix);
      int iy = (int)floorf(cy); iy = iy < 0 ? 0 : (iy > GRES - 1 ? GRES - 1 : iy);
      int iz = (int)floorf(cz); iz = iz < 0 ? 0 : (iz > GRES - 1 ? GRES - 1 : iz);
      bool occv = occ[(ix * GRES + iy) * GRES + iz] != 0;
      float validf = ((tval < tf) && inbox && occv) ? 1.0f : 0.0f;
      sm.tv[s]  = tval;
      sm.vld[s] = validf;
      _Float16* pr = sm.u.pe[wave][lane];
      pr[0] = (_Float16)px; pr[1] = (_Float16)py; pr[2] = (_Float16)pz;
      float p3[3] = {px, py, pz};
#pragma unroll
      for (int i = 0; i < 3; ++i) {
#pragma unroll
        for (int l = 0; l < 4; ++l) {
          float a = p3[i] * (float)(1 << l);
          pr[3 + i * 8 + l]     = (_Float16)__sinf(a);
          pr[3 + i * 8 + 4 + l] = (_Float16)__cosf(a);
        }
      }
#pragma unroll
      for (int k = 27; k < 32; ++k) pr[k] = (_Float16)0.0f;
    }
    WAVE_SYNC();  // wave-private staging; LDS in-order within wave

    // ---- Phase B: layer 1 (K=32, N=64) ----
    {
      v16h a = lds_load_A(&sm.u.pe[wave][0][0], 32, 0, lane);
      int col = lane & 15, mb = (lane & 16) ? 8 : 0;
#pragma unroll
      for (int t = 0; t < 4; ++t) {
        v16h bf = lds_load_B(&sm.W1t[0][0], 32, t * 16, 0, lane);
        v8f c = {};
        c = WMMA_F16(a, bf, c);
        int j = t * 16 + col;
        float bj = sm.b1f[j];
#pragma unroll
        for (int r = 0; r < 8; ++r)
          sm.h1[wave][mb + r][j] = (_Float16)fmaxf(c[r] + bj, 0.0f);
      }
    }
    WAVE_SYNC();

    // ---- Phase C: layer 2 (K=64, N=64) ----
    {
      v16h a0 = lds_load_A(&sm.h1[wave][0][0], 64, 0, lane);
      v16h a1 = lds_load_A(&sm.h1[wave][0][0], 64, 32, lane);
      int col = lane & 15, mb = (lane & 16) ? 8 : 0;
#pragma unroll
      for (int t = 0; t < 4; ++t) {
        v16h b0 = lds_load_B(&sm.W2t[0][0], 64, t * 16, 0, lane);
        v16h b1f_ = lds_load_B(&sm.W2t[0][0], 64, t * 16, 32, lane);
        v8f c = {};
        c = WMMA_F16(a0, b0, c);
        c = WMMA_F16(a1, b1f_, c);
        int j = t * 16 + col;
        float bj = sm.b2f[j];
#pragma unroll
        for (int r = 0; r < 8; ++r)
          sm.u.h2[wave][mb + r][j] = (_Float16)fmaxf(c[r] + bj, 0.0f);
      }
    }
    WAVE_SYNC();

    // ---- Phase D: sigma head (WMMA, padded N-tile) + color layer g ----
    {
      v16h a0 = lds_load_A(&sm.u.h2[wave][0][0], 64, 0, lane);
      v16h a1 = lds_load_A(&sm.u.h2[wave][0][0], 64, 32, lane);
      int col = lane & 15, mb = (lane & 16) ? 8 : 0;
      {
        v16h b0 = lds_load_B(&sm.Wst[0][0], 64, 0, 0, lane);
        v16h b1f_ = lds_load_B(&sm.Wst[0][0], 64, 0, 32, lane);
        v8f c = {};
        c = WMMA_F16(a0, b0, c);
        c = WMMA_F16(a1, b1f_, c);
        if (col == 0) {
#pragma unroll
          for (int r = 0; r < 8; ++r) {
            int s = wave * 16 + mb + r;
            sm.sig[s] = fmaxf(c[r] + bsv, 0.0f) * sm.vld[s];
          }
        }
      }
#pragma unroll
      for (int t = 0; t < 4; ++t) {
        v16h b0 = lds_load_B(&sm.Wct[0][0], 64, t * 16, 0, lane);
        v16h b1f_ = lds_load_B(&sm.Wct[0][0], 64, t * 16, 32, lane);
        v8f c = {};
        c = WMMA_F16(a0, b0, c);
        c = WMMA_F16(a1, b1f_, c);
        int j = t * 16 + col;
        float dj = sm.dfeat[j];
#pragma unroll
        for (int r = 0; r < 8; ++r)
          sm.h1[wave][mb + r][j] = (_Float16)fmaxf(c[r] + dj, 0.0f);  // g reuses h1
      }
    }
    WAVE_SYNC();

    // ---- Phase E: rgb head (WMMA, padded N-tile, cols 0..2 live) ----
    {
      v16h a0 = lds_load_A(&sm.h1[wave][0][0], 64, 0, lane);
      v16h a1 = lds_load_A(&sm.h1[wave][0][0], 64, 32, lane);
      v16h b0 = lds_load_B(&sm.Wrt[0][0], 64, 0, 0, lane);
      v16h b1f_ = lds_load_B(&sm.Wrt[0][0], 64, 0, 32, lane);
      v8f c = {};
      c = WMMA_F16(a0, b0, c);
      c = WMMA_F16(a1, b1f_, c);
      int col = lane & 15, mb = (lane & 16) ? 8 : 0;
      if (col < 3) {
        float bb = br[col];
#pragma unroll
        for (int r = 0; r < 8; ++r) {
          int s = wave * 16 + mb + r;
          float x = c[r] + bb;
          sm.rgbb[s][col] = 1.0f / (1.0f + __expf(-x));
        }
      }
    }
    __syncthreads();  // cross-wave: wave 0 consumes all waves' sig/tv/rgbb

    // ---- Phase F: alpha compositing (wave 0, 4 samples/lane, wave prefix scan) ----
    if (wave == 0) {
      float sd[4], pre[4];
      float run = 0.0f;
#pragma unroll
      for (int q = 0; q < 4; ++q) {
        sd[q] = sm.sig[lane * 4 + q] * STEP_F;
        run += sd[q];
        pre[q] = run;  // inclusive within-lane
      }
      float tot = run, scan = tot;
#pragma unroll
      for (int off = 1; off < 32; off <<= 1) {
        float v = __shfl_up(scan, off, 32);
        if (lane >= off) scan += v;
      }
      float excl = scan - tot;  // exclusive lane prefix
      float r0 = 0.f, r1 = 0.f, r2 = 0.f, aacc = 0.f, dacc = 0.f;
#pragma unroll
      for (int q = 0; q < 4; ++q) {
        int s = lane * 4 + q;
        float cum = excl + pre[q];
        float T = __expf(-(cum - sd[q]));
        float alpha = 1.0f - __expf(-sd[q]);
        float w = T * alpha;
        r0 += w * sm.rgbb[s][0];
        r1 += w * sm.rgbb[s][1];
        r2 += w * sm.rgbb[s][2];
        aacc += w;
        dacc += w * sm.tv[s];
      }
#pragma unroll
      for (int off = 16; off > 0; off >>= 1) {
        r0 += __shfl_down(r0, off, 32);
        r1 += __shfl_down(r1, off, 32);
        r2 += __shfl_down(r2, off, 32);
        aacc += __shfl_down(aacc, off, 32);
        dacc += __shfl_down(dacc, off, 32);
      }
      if (lane == 0) {
        float bg = 1.0f - aacc;
        out[ray * 3 + 0] = r0 + bg;
        out[ray * 3 + 1] = r1 + bg;
        out[ray * 3 + 2] = r2 + bg;
        out[NRAYS * 3 + ray] = dacc;           // depth
        out[NRAYS * 4 + ray] = aacc;           // acc
      }
    }
    __syncthreads();  // cross-wave: staging buffers reused next ray
  }
}

extern "C" void kernel_launch(void* const* d_in, const int* in_sizes, int n_in,
                              void* d_out, int out_size, void* d_ws, size_t ws_size,
                              hipStream_t stream) {
  const float* rays_o   = (const float*)d_in[0];
  const float* viewdirs = (const float*)d_in[1];
  const unsigned char* occ = (const unsigned char*)d_in[2];  // jax bool = 1 byte/elem
  const float* W1 = (const float*)d_in[3];
  const float* b1 = (const float*)d_in[4];
  const float* W2 = (const float*)d_in[5];
  const float* b2 = (const float*)d_in[6];
  const float* Ws = (const float*)d_in[7];
  const float* bs = (const float*)d_in[8];
  const float* Wc = (const float*)d_in[9];
  const float* Wd = (const float*)d_in[10];
  const float* bc = (const float*)d_in[11];
  const float* Wr = (const float*)d_in[12];
  const float* br = (const float*)d_in[13];
  float* out = (float*)d_out;
  (void)in_sizes; (void)n_in; (void)out_size; (void)d_ws; (void)ws_size;

  nerf_render_kernel<<<dim3(NBLK), dim3(256), 0, stream>>>(
      rays_o, viewdirs, occ, W1, b1, W2, b2, Ws, bs, Wc, Wd, bc, Wr, br, out);
}
